// Interface_47072841564867
// MI455X (gfx1250) — matrix operand order repaired and checked by reading, not compiled, not run on hardware
//
#include <hip/hip_runtime.h>
#include <hip/hip_bf16.h>

#define N_NN 16
#define N_NI 64
#define N_ND 128

typedef __attribute__((ext_vector_type(16))) _Float16 v16h;
typedef __attribute__((ext_vector_type(8)))  _Float16 v8h;
typedef __attribute__((ext_vector_type(8)))  float    v8f;

// ---------------------------------------------------------------------------
// Pre-pass: convert W1 (128x128) and W2 (128x128) f32 row-major (K x N) into
// f16 transposed (N x K) so WMMA B-fragments are contiguous 16B loads.
// ---------------------------------------------------------------------------
__global__ __launch_bounds__(256)
void cvt_weights_kernel(const float* __restrict__ W1, const float* __restrict__ W2,
                        _Float16* __restrict__ Wt1, _Float16* __restrict__ Wt2)
{
    int e = blockIdx.x * blockDim.x + threadIdx.x;
    if (e < N_ND * N_ND) {
        int n = e >> 7, k = e & 127;
        Wt1[n * N_ND + k] = (_Float16)W1[k * N_ND + n];
    } else if (e < 2 * N_ND * N_ND) {
        int e2 = e - N_ND * N_ND;
        int n = e2 >> 7, k = e2 & 127;
        Wt2[n * N_ND + k] = (_Float16)W2[k * N_ND + n];
    }
}

// ---------------------------------------------------------------------------
// Fragment loaders (wave32 WMMA layouts per CDNA5 ISA 7.12.2)
// ---------------------------------------------------------------------------
// A (16x32 f16): lane<16 -> row=lane, K = kb+{0..7} and kb+16+{0..7}
//                lane>=16 -> row=lane-16, K = kb+8+{0..7} and kb+24+{0..7}
__device__ __forceinline__ v16h load_a_frag(const _Float16* __restrict__ h,
                                            int kbase, int lane)
{
    const int row = lane & 15;
    const int ko  = (lane < 16) ? 0 : 8;
    const v8h lo = *(const v8h*)(h + row * N_ND + kbase + ko);
    const v8h hi = *(const v8h*)(h + row * N_ND + kbase + 16 + ko);
    v16h a;
#pragma unroll
    for (int i = 0; i < 8; ++i) { a[i] = lo[i]; a[i + 8] = hi[i]; }
    return a;
}

// B (32x16 f16), weights stored N-major (col c is contiguous over K):
// lane<16 -> col=ncol+lane, K = kb+{0..7} and kb+16+{0..7} ; lane>=16 -> col=ncol+lane-16, K = kb+8+{0..7} and kb+24+{0..7}
__device__ __forceinline__ v16h load_b_frag(const _Float16* __restrict__ Wt,
                                            int ncol, int kbase, int lane)
{
    const int c  = ncol + (lane & 15);
    const int ko = (lane < 16) ? 0 : 8;
    const v8h lo = *(const v8h*)(Wt + c * N_ND + kbase + ko);
    const v8h hi = *(const v8h*)(Wt + c * N_ND + kbase + ko + 16);
    v16h b;
#pragma unroll
    for (int i = 0; i < 8; ++i) { b[i] = lo[i]; b[i + 8] = hi[i]; }
    return b;
}

__device__ __forceinline__ float elu_f(float x)
{
    return x > 0.0f ? x : (__expf(x) - 1.0f);
}

// ---------------------------------------------------------------------------
// Persistent fused kernel: fixed grid, each block (4 waves) strides over
// queries. Weight fragments live in registers for the whole kernel; all four
// A-fragments are prefetched per GEMM so LDS latency overlaps WMMA issue.
// ---------------------------------------------------------------------------
__global__ __launch_bounds__(128)
void fused_knn_mlp_kernel(const float* __restrict__ f1, const float* __restrict__ f2,
                          const float* __restrict__ x1, const float* __restrict__ x2,
                          const long long* __restrict__ topk,
                          const _Float16* __restrict__ Wt1, const _Float16* __restrict__ Wt2,
                          const float* __restrict__ b1v, const float* __restrict__ b2v,
                          const float* __restrict__ radius_p,
                          float* __restrict__ out, int n1)
{
    __shared__ __align__(16) _Float16 sh0[N_NN * N_ND];   // input tile  16x128 f16
    __shared__ __align__(16) _Float16 sh1[N_NN * N_ND];   // hidden tile 16x128 f16
    __shared__ float swt[N_NN];                            // neighbor weights

    const int tid  = threadIdx.x;
    const int lane = tid & 31;
    const int wv   = tid >> 5;            // wave 0..3 -> columns [wv*32, wv*32+32)

    // ---- loop-invariant hoists: register-resident weight fragments ----
    v16h w1f[2][4], w2f[2][4];
#pragma unroll
    for (int t = 0; t < 2; ++t)
#pragma unroll
        for (int kt = 0; kt < 4; ++kt) {
            w1f[t][kt] = load_b_frag(Wt1, wv * 32 + t * 16, kt * 32, lane);
            w2f[t][kt] = load_b_frag(Wt2, wv * 32 + t * 16, kt * 32, lane);
        }

    const int   c0   = wv * 32 + (lane & 15);
    const int   c1   = c0 + 16;
    const float bia0 = b1v[c0], bia1 = b1v[c1];
    const float bib0 = b2v[c0], bib1 = b2v[c1];
    const float rad  = radius_p[0];
    const float inv2r2 = 1.0f / (2.0f * rad * rad);
    const int   mbase  = (lane < 16) ? 0 : 8;   // C layout: VGPR r -> row mbase+r

    for (int q = blockIdx.x; q < n1; q += gridDim.x) {
        // ---- stage h0 = [features2[topk] | features1 broadcast] as f16 ----
        {
            const int row = tid >> 3;         // neighbor 0..15
            const int seg = tid & 7;          // 16-column segment
            const long long nb = topk[(long long)q * N_NN + row];
            const int coff = seg * 16;
            const float* src = (seg < 4) ? (f2 + (long long)nb * N_NI + coff)
                                         : (f1 + (long long)q * N_NI + (coff - 64));
            _Float16* dst = sh0 + row * N_ND + coff;
#pragma unroll
            for (int i = 0; i < 16; i += 4) {
                float4 v = *(const float4*)(src + i);
                dst[i + 0] = (_Float16)v.x; dst[i + 1] = (_Float16)v.y;
                dst[i + 2] = (_Float16)v.z; dst[i + 3] = (_Float16)v.w;
            }
        }
        if (tid < N_NN) {
            const long long nb = topk[(long long)q * N_NN + tid];
            const float dx = x2[nb * 3 + 0] - x1[q * 3 + 0];
            const float dy = x2[nb * 3 + 1] - x1[q * 3 + 1];
            const float dz = x2[nb * 3 + 2] - x1[q * 3 + 2];
            const float d2 = dx * dx + dy * dy + dz * dz;
            const float w  = __expf(-d2 * inv2r2);
            swt[tid] = (nb == 0) ? 0.0f : w;
        }
        __syncthreads();

        // ---- GEMM1: prefetch all A-fragments, then 8 WMMAs ----
        {
            const v16h a0 = load_a_frag(sh0,  0, lane);
            const v16h a1 = load_a_frag(sh0, 32, lane);
            const v16h a2 = load_a_frag(sh0, 64, lane);
            const v16h a3 = load_a_frag(sh0, 96, lane);
            v8f acc0 = {}, acc1 = {};
            acc0 = __builtin_amdgcn_wmma_f32_16x16x32_f16(false, a0, false, w1f[0][0], (short)0, acc0, false, false);
            acc1 = __builtin_amdgcn_wmma_f32_16x16x32_f16(false, a0, false, w1f[1][0], (short)0, acc1, false, false);
            acc0 = __builtin_amdgcn_wmma_f32_16x16x32_f16(false, a1, false, w1f[0][1], (short)0, acc0, false, false);
            acc1 = __builtin_amdgcn_wmma_f32_16x16x32_f16(false, a1, false, w1f[1][1], (short)0, acc1, false, false);
            acc0 = __builtin_amdgcn_wmma_f32_16x16x32_f16(false, a2, false, w1f[0][2], (short)0, acc0, false, false);
            acc1 = __builtin_amdgcn_wmma_f32_16x16x32_f16(false, a2, false, w1f[1][2], (short)0, acc1, false, false);
            acc0 = __builtin_amdgcn_wmma_f32_16x16x32_f16(false, a3, false, w1f[0][3], (short)0, acc0, false, false);
            acc1 = __builtin_amdgcn_wmma_f32_16x16x32_f16(false, a3, false, w1f[1][3], (short)0, acc1, false, false);
#pragma unroll
            for (int r = 0; r < 8; ++r) {
                const int m = mbase + r;
                sh1[m * N_ND + c0] = (_Float16)elu_f(acc0[r] + bia0);
                sh1[m * N_ND + c1] = (_Float16)elu_f(acc1[r] + bia1);
            }
        }
        __syncthreads();

        // ---- GEMM2 + ELU + distance-weighted neighbor reduction ----
        {
            const v16h a0 = load_a_frag(sh1,  0, lane);
            const v16h a1 = load_a_frag(sh1, 32, lane);
            const v16h a2 = load_a_frag(sh1, 64, lane);
            const v16h a3 = load_a_frag(sh1, 96, lane);
            v8f d0 = {}, d1 = {};
            d0 = __builtin_amdgcn_wmma_f32_16x16x32_f16(false, a0, false, w2f[0][0], (short)0, d0, false, false);
            d1 = __builtin_amdgcn_wmma_f32_16x16x32_f16(false, a0, false, w2f[1][0], (short)0, d1, false, false);
            d0 = __builtin_amdgcn_wmma_f32_16x16x32_f16(false, a1, false, w2f[0][1], (short)0, d0, false, false);
            d1 = __builtin_amdgcn_wmma_f32_16x16x32_f16(false, a1, false, w2f[1][1], (short)0, d1, false, false);
            d0 = __builtin_amdgcn_wmma_f32_16x16x32_f16(false, a2, false, w2f[0][2], (short)0, d0, false, false);
            d1 = __builtin_amdgcn_wmma_f32_16x16x32_f16(false, a2, false, w2f[1][2], (short)0, d1, false, false);
            d0 = __builtin_amdgcn_wmma_f32_16x16x32_f16(false, a3, false, w2f[0][3], (short)0, d0, false, false);
            d1 = __builtin_amdgcn_wmma_f32_16x16x32_f16(false, a3, false, w2f[1][3], (short)0, d1, false, false);

            float s0 = 0.0f, s1 = 0.0f;
#pragma unroll
            for (int r = 0; r < 8; ++r) {
                const float wk = swt[mbase + r];
                s0 += wk * elu_f(d0[r] + bib0);
                s1 += wk * elu_f(d1[r] + bib1);
            }
            // rows 0-7 live in lanes 0-15, rows 8-15 in lanes 16-31 (same column)
            s0 += __shfl_xor(s0, 16, 32);
            s1 += __shfl_xor(s1, 16, 32);
            if (lane < 16) {
                out[(long long)q * N_ND + c0] = s0;
                out[(long long)q * N_ND + c1] = s1;
            }
        }
        __syncthreads();   // protect sh0/swt reuse next iteration
    }
}

extern "C" void kernel_launch(void* const* d_in, const int* in_sizes, int n_in,
                              void* d_out, int out_size, void* d_ws, size_t ws_size,
                              hipStream_t stream)
{
    const float*     f1   = (const float*)d_in[0];
    const float*     f2   = (const float*)d_in[1];
    const float*     x1   = (const float*)d_in[2];
    const float*     x2   = (const float*)d_in[3];
    // d_in[4], d_in[5] (nuv1/nuv2) unused by the reference output
    const long long* topk = (const long long*)d_in[6];
    const float*     W1   = (const float*)d_in[7];
    const float*     b1   = (const float*)d_in[8];
    const float*     W2   = (const float*)d_in[9];
    const float*     b2   = (const float*)d_in[10];
    const float*     rad  = (const float*)d_in[11];
    float*           out  = (float*)d_out;

    const int n1 = in_sizes[0] / N_NI;   // number of query points

    _Float16* Wt1 = (_Float16*)d_ws;
    _Float16* Wt2 = Wt1 + N_ND * N_ND;

    // weights -> f16, transposed (runs every call; deterministic)
    cvt_weights_kernel<<<(2 * N_ND * N_ND + 255) / 256, 256, 0, stream>>>(W1, W2, Wt1, Wt2);

    const int grid = (n1 < 2048) ? n1 : 2048;
    fused_knn_mlp_kernel<<<grid, 128, 0, stream>>>(f1, f2, x1, x2, topk,
                                                   Wt1, Wt2, b1, b2, rad, out, n1);
}
